// ProteinMHAttentionSummarizer_82154134438753
// MI455X (gfx1250) — matrix operand order repaired and checked by reading, not compiled, not run on hardware
//
#include <hip/hip_runtime.h>

#define B_  32
#define C_  512
#define L_  4096
#define H_  8
#define WIN 9
#define PADW 4

typedef __attribute__((ext_vector_type(16))) __bf16         v16bf;
typedef __attribute__((ext_vector_type(16))) unsigned short v16u;
typedef __attribute__((ext_vector_type(8)))  float          v8f;
typedef __attribute__((ext_vector_type(4)))  unsigned int   u32x4;
typedef __attribute__((ext_vector_type(8)))  int            i32x8;
typedef __attribute__((ext_vector_type(4)))  int            i32x4;

__device__ __forceinline__ unsigned short f2bf(float f) {
  union { float f; unsigned u; } v; v.f = f;
  unsigned r = v.u + 0x7FFFu + ((v.u >> 16) & 1u);   // round-to-nearest-even
  return (unsigned short)(r >> 16);
}

// 1-D TDM copy: nelem bf16 elements global -> LDS (wave-level op, EXEC-ignored).
// D# per CDNA5 ISA ch.8: group0 = {count, lds_addr, global_addr, type=2},
// group1 = {data_size=1(2B), tensor_dim0=tile_dim0=nelem, dim1=1, stride=nelem}.
// 6-arg builtin form (clang-23 toolchain): (g0, g1, g2, g3, g4, cpol).
__device__ __forceinline__ void tdm_load_1d(unsigned lds_off, const void* gptr,
                                            unsigned nelem) {
  unsigned long long ga = (unsigned long long)(size_t)gptr;
  u32x4 g0;
  g0[0] = 1u;                                           // count=1 (valid user D#)
  g0[1] = lds_off;                                      // lds_addr (bytes)
  g0[2] = (unsigned)(ga & 0xFFFFFFFFu);                 // global_addr[31:0]
  g0[3] = (unsigned)((ga >> 32) & 0x1FFFFFFu) | (2u << 30);  // ga[56:32] | type=2
  i32x8 g1;
  g1[0] = (int)(1u << 16);                              // data_size=1 -> 2 bytes
  g1[1] = (int)((nelem & 0xFFFFu) << 16);               // tensor_dim0[15:0]
  g1[2] = (int)((nelem >> 16) | (1u << 16));            // tensor_dim0[31:16] | dim1=1
  g1[3] = (int)((nelem & 0xFFFFu) << 16);               // dim1 hi=0 | tile_dim0
  g1[4] = 1;                                            // tile_dim1=1, tile_dim2=0
  g1[5] = (int)nelem;                                   // tensor_dim0_stride lo
  g1[6] = 0;                                            // stride hi | dim1_stride lo
  g1[7] = 0;
  i32x4 z4 = {0, 0, 0, 0};
  i32x8 z8 = {0, 0, 0, 0, 0, 0, 0, 0};
  __builtin_amdgcn_tensor_load_to_lds(g0, g1, z4, z4, z8, 0);
}

// ---------------------------------------------------------------------------
// Kernel 1: pre-pack W [H,C,WIN] f32 into bf16 WMMA A-fragments.
// Fragment (w, cc) holds A[m=h][k] in the 16-bit A lane layout:
// lane = khalf*16 + h ; element j -> K = j + 16*khalf (contiguous k order; B uses the same).
// ---------------------------------------------------------------------------
__global__ void prep_wfrag(const float* __restrict__ W, unsigned short* __restrict__ wf) {
  int blk  = blockIdx.x;            // w*16 + cc
  int w    = blk / 16;
  int cc   = blk % 16;
  int lane = threadIdx.x;
  int h     = lane & 15;
  int khalf = lane >> 4;
  v16u out = {};
  if (h < H_) {
#pragma unroll
    for (int j = 0; j < 16; ++j) {
      int K = j + 16 * khalf;
      int c = cc * 32 + K;
      out[j] = f2bf(W[((size_t)h * C_ + c) * WIN + w]);
    }
  }
  *(v16u*)(wf + ((size_t)blk * 32 + lane) * 16) = out;
}

// ---------------------------------------------------------------------------
// Kernel 2: transpose x [b][c][l] f32 -> xT [b][l][c] bf16 (L2-resident).
// ---------------------------------------------------------------------------
__global__ void __launch_bounds__(256) transpose_x(const float* __restrict__ x,
                                                   unsigned short* __restrict__ xT) {
  __shared__ float t[32][33];
  int lt = blockIdx.x, ct = blockIdx.y, b = blockIdx.z;
  int tx = threadIdx.x & 31;
  int ty = threadIdx.x >> 5;
  const float* xb = x + ((size_t)b * C_ + ct * 32) * L_ + (size_t)lt * 32;
#pragma unroll
  for (int i = 0; i < 4; ++i) {
    int c = ty + 8 * i;
    t[c][tx] = xb[(size_t)c * L_ + tx];
  }
  __syncthreads();
  unsigned short* o = xT + ((size_t)b * L_ + lt * 32) * C_ + ct * 32;
#pragma unroll
  for (int i = 0; i < 4; ++i) {
    int l = ty + 8 * i;
    o[(size_t)l * C_ + tx] = f2bf(t[tx][l]);
  }
}

// ---------------------------------------------------------------------------
// Kernel 3: conv1d as WMMA GEMM, A-fragments staged by TDM (double-buffered).
// D tile = [16 heads x 16 l].  B[k][n] = xT[b][l0+n+w-4][cc*32+k] : per lane a
// contiguous 32B row segment of xT.  A read from LDS after tensor_load_to_lds.
// ---------------------------------------------------------------------------
__global__ void __launch_bounds__(128) conv_wmma(const unsigned short* __restrict__ xT,
                                                 const unsigned short* __restrict__ wf,
                                                 float* __restrict__ logits) {
  __shared__ __align__(32) unsigned short awbuf[2][16 * 32 * 16];  // 2 x 16 KB
  int b    = blockIdx.y;
  int wave = threadIdx.x >> 5;
  int lane = threadIdx.x & 31;
  int l0   = blockIdx.x * 64 + wave * 16;
  int n     = lane & 15;
  int khalf = lane >> 4;
  unsigned lds0 = (unsigned)(size_t)(&awbuf[0][0]);   // flat addr low 32b == LDS offset
  v8f acc = {};
  const unsigned short* xb = xT + (size_t)b * L_ * C_;

  if (wave == 0) tdm_load_1d(lds0, wf, 16 * 32 * 16);           // tap 0 -> buf 0

#pragma unroll 1
  for (int w = 0; w < WIN; ++w) {
    if (wave == 0) {
      if (w + 1 < WIN) {
        tdm_load_1d(lds0 + (unsigned)(((w + 1) & 1) * 16 * 32 * 16 * 2),
                    wf + (size_t)(w + 1) * 16 * 32 * 16, 16 * 32 * 16);
        __builtin_amdgcn_s_wait_tensorcnt(1);   // tap w resident; w+1 in flight
      } else {
        __builtin_amdgcn_s_wait_tensorcnt(0);
      }
    }
    __syncthreads();

    int  lrow = l0 + n + w - PADW;
    bool inb  = (lrow >= 0) && (lrow < L_);
    const unsigned short* rp = xb + (size_t)lrow * C_ + khalf * 16;
    const unsigned short* ap = &awbuf[w & 1][lane * 16];
    if (inb && (w + 1 < WIN) && (lrow + 1 < L_))
      __builtin_prefetch(rp + C_, 0, 3);        // global_prefetch_b8 of next tap row
#pragma unroll 8
    for (int cc = 0; cc < 16; ++cc) {
      v16u braw = {};
      if (inb) braw = *(const v16u*)(rp + cc * 32);
      v16u araw = *(const v16u*)(ap + cc * 512);          // ds_load from staged frags
      v16bf av = __builtin_bit_cast(v16bf, araw);
      v16bf bv = __builtin_bit_cast(v16bf, braw);
      acc = __builtin_amdgcn_wmma_f32_16x16x32_bf16(false, av, false, bv,
                                                    (short)0, acc, false, false);
    }
    __syncthreads();                            // buf (w&1) free for tap w+2
  }
  if (lane < 16) {
    float* lg = logits + (size_t)b * H_ * L_ + l0 + lane;
#pragma unroll
    for (int r = 0; r < H_; ++r) lg[(size_t)r * L_] = acc[r];
  }
}

// ---------------------------------------------------------------------------
// Kernel 4: softmax over L per (b,h) row; output bf16 focus.
// (bias b[h] is a constant shift per row -> softmax-invariant -> omitted)
// ---------------------------------------------------------------------------
__global__ void __launch_bounds__(256) softmax_rows(const float* __restrict__ logits,
                                                    unsigned short* __restrict__ focus) {
  __shared__ float red[256];
  int row = blockIdx.x;
  const float* lr = logits + (size_t)row * L_;
  int t = threadIdx.x;
  float mx = -3.0e38f;
  for (int i = t; i < L_; i += 256) mx = fmaxf(mx, lr[i]);
  red[t] = mx; __syncthreads();
  for (int s = 128; s > 0; s >>= 1) { if (t < s) red[t] = fmaxf(red[t], red[t + s]); __syncthreads(); }
  mx = red[0]; __syncthreads();
  float sum = 0.f;
  for (int i = t; i < L_; i += 256) sum += __expf(lr[i] - mx);
  red[t] = sum; __syncthreads();
  for (int s = 128; s > 0; s >>= 1) { if (t < s) red[t] += red[t + s]; __syncthreads(); }
  float inv = 1.0f / red[0];
  unsigned short* fr = focus + (size_t)row * L_;
  for (int i = t; i < L_; i += 256) fr[i] = f2bf(__expf(lr[i] - mx) * inv);
}

// ---------------------------------------------------------------------------
// Kernel 5: pooling GEMM + head-max.  Per wave: D tile [16 c x 16 h].
// ---------------------------------------------------------------------------
__global__ void __launch_bounds__(128) pool_wmma(const float* __restrict__ x,
                                                 const unsigned short* __restrict__ focus,
                                                 float* __restrict__ out) {
  int b    = blockIdx.y;
  int wave = threadIdx.x >> 5;
  int lane = threadIdx.x & 31;
  int c0   = blockIdx.x * 64 + wave * 16;
  int m     = lane & 15;
  int khalf = lane >> 4;
  v8f acc = {};
  const float* xr = x + ((size_t)b * C_ + c0 + m) * L_;
  const unsigned short* fr =
      (m < H_) ? (focus + ((size_t)b * H_ + m) * L_ + khalf * 16) : (const unsigned short*)0;
#pragma unroll 2
  for (int l0 = 0; l0 < L_; l0 += 32) {
    const float4* q0 = (const float4*)(xr + l0 + 16 * khalf);
    const float4* q1 = (const float4*)(xr + l0 + 16 * khalf + 8);
    float4 A0 = q0[0], A1 = q0[1];
    float4 A2 = q1[0], A3 = q1[1];
    v16u araw;
    araw[0]  = f2bf(A0.x); araw[1]  = f2bf(A0.y); araw[2]  = f2bf(A0.z); araw[3]  = f2bf(A0.w);
    araw[4]  = f2bf(A1.x); araw[5]  = f2bf(A1.y); araw[6]  = f2bf(A1.z); araw[7]  = f2bf(A1.w);
    araw[8]  = f2bf(A2.x); araw[9]  = f2bf(A2.y); araw[10] = f2bf(A2.z); araw[11] = f2bf(A2.w);
    araw[12] = f2bf(A3.x); araw[13] = f2bf(A3.y); araw[14] = f2bf(A3.z); araw[15] = f2bf(A3.w);
    v16u braw = {};
    if (fr) braw = *(const v16u*)(fr + l0);
    v16bf av = __builtin_bit_cast(v16bf, araw);
    v16bf bv = __builtin_bit_cast(v16bf, braw);
    acc = __builtin_amdgcn_wmma_f32_16x16x32_bf16(false, av, false, bv,
                                                  (short)0, acc, false, false);
  }
#pragma unroll
  for (int r = 0; r < 8; ++r) {
    float v = acc[r];
    v = fmaxf(v, __shfl_xor(v, 1, 32));
    v = fmaxf(v, __shfl_xor(v, 2, 32));
    v = fmaxf(v, __shfl_xor(v, 4, 32));
    acc[r] = v;
  }
  if ((lane & 15) == 0) {
    float* op = out + (size_t)b * C_ + c0 + (khalf ? 8 : 0);
    *(float4*)(op)     = make_float4(acc[0], acc[1], acc[2], acc[3]);
    *(float4*)(op + 4) = make_float4(acc[4], acc[5], acc[6], acc[7]);
  }
}

// ---------------------------------------------------------------------------
extern "C" void kernel_launch(void* const* d_in, const int* in_sizes, int n_in,
                              void* d_out, int out_size, void* d_ws, size_t ws_size,
                              hipStream_t stream) {
  const float* x = (const float*)d_in[0];   // [B,C,L] f32
  const float* W = (const float*)d_in[1];   // [H,C,WIN] f32
  // d_in[2] = bias b[H]: constant per softmax row -> output-invariant, unused.
  float* out = (float*)d_out;               // [B,C] f32
  (void)in_sizes; (void)n_in; (void)out_size; (void)ws_size;

  char* ws = (char*)d_ws;
  size_t off = 0;
  unsigned short* wfrag = (unsigned short*)(ws + off);
  off += (size_t)WIN * 16 * 32 * 16 * 2;  off = (off + 255) & ~(size_t)255;
  float* logits = (float*)(ws + off);
  off += (size_t)B_ * H_ * L_ * 4;        off = (off + 255) & ~(size_t)255;
  unsigned short* focus = (unsigned short*)(ws + off);
  off += (size_t)B_ * H_ * L_ * 2;        off = (off + 255) & ~(size_t)255;
  unsigned short* xT = (unsigned short*)(ws + off);

  prep_wfrag   <<<dim3(WIN * 16),              dim3(32),  0, stream>>>(W, wfrag);
  transpose_x  <<<dim3(L_ / 32, C_ / 32, B_),  dim3(256), 0, stream>>>(x, xT);
  conv_wmma    <<<dim3(L_ / 64, B_),           dim3(128), 0, stream>>>(xT, wfrag, logits);
  softmax_rows <<<dim3(B_ * H_),               dim3(256), 0, stream>>>(logits, focus);
  pool_wmma    <<<dim3(C_ / 64, B_),           dim3(128), 0, stream>>>(x, focus, out);
}
